// Model2_13683765805466
// MI455X (gfx1250) — matrix operand order repaired and checked by reading, not compiled, not run on hardware
//
#include <hip/hip_runtime.h>
#include <hip/hip_bf16.h>

// ---------------------------------------------------------------------------
// Shapes (compile-time constants from the reference)
// ---------------------------------------------------------------------------
#define D_    256
#define T_    16
#define B_    512
#define L_    16
#define NTGT_ 32

typedef __attribute__((ext_vector_type(16))) __bf16 v16bf;
typedef __attribute__((ext_vector_type(8)))  float  v8f;

union Frag32 { uint4 u[2]; v16bf v; };   // 32 bytes = 16 bf16 = one WMMA operand/lane

// ---------------------------------------------------------------------------
// K0: gather item embeddings for the sequence -> f32 (for fo-pool x) + bf16 (GEMM A)
// grid = B*T blocks, block = D threads
// ---------------------------------------------------------------------------
__global__ void gather_emb_kernel(const int* __restrict__ seq,
                                  const float* __restrict__ item_emb,
                                  float* __restrict__ emb_f32,
                                  __hip_bfloat16* __restrict__ emb_bf16) {
  int r = blockIdx.x;          // b*T + t
  int d = threadIdx.x;
  int row = seq[r];
  float v = item_emb[(size_t)row * D_ + d];
  emb_f32[(size_t)r * D_ + d]  = v;
  emb_bf16[(size_t)r * D_ + d] = __float2bfloat16(v);
}

// ---------------------------------------------------------------------------
// K1: repack conv_w[l][m][c][d] (f32) -> Bt[(l*D+c)][(m*D+d)] (bf16)
// so each GEMM column's K axis (m,d) is contiguous.  grid = (L*L, D), block = D
// ---------------------------------------------------------------------------
__global__ void pack_w_kernel(const float* __restrict__ conv_w,
                              __hip_bfloat16* __restrict__ Bt) {
  int lm = blockIdx.x;               // l*16 + m
  int c  = blockIdx.y;
  int d  = threadIdx.x;
  int l = lm >> 4, m = lm & 15;
  float v = conv_w[((size_t)lm * D_ + c) * D_ + d];
  Bt[((size_t)(l * D_ + c)) * (L_ * D_) + m * D_ + d] = __float2bfloat16(v);
}

// ---------------------------------------------------------------------------
// K2: causal-conv GEMM with v_wmma_f32_16x16x32_bf16 + fused relu->sigmoid gate.
//   conv_out[l,b,t,c] = sum_{m<=l} sum_d emb[b,t-m,d] * conv_w[l,m,c,d] + conv_b[l,c]
// One WG per (b, 512-column slab).  8 waves; wave owns 4 C tiles (64 cols) in one
// l-block => uniform m-loop (triangular skip).  `l` is forced into an SGPR via
// readfirstlane so the K-loop and prefetch guard compile to scalar branches and
// EXEC is structurally all-ones around every WMMA.
// A tiles come from LDS (16 emb rows + 1 zero row for the causal pad).
// Two-deep register ping-pong pipelines the B-fragment global loads past the
// WMMAs so the backend issues partial s_wait_loadcnt instead of draining.
// ---------------------------------------------------------------------------
__global__ void __launch_bounds__(256)
conv_gemm_kernel(const __hip_bfloat16* __restrict__ emb_bf16,
                 const __hip_bfloat16* __restrict__ Bt,
                 const float* __restrict__ conv_b,
                 __hip_bfloat16* __restrict__ fbuf) {
  __shared__ uint4 lds[(17 * D_ * 2) / 16];         // 16 emb rows + 1 zero row (8704 B)

  const int b     = blockIdx.y;
  const int xslab = blockIdx.x;                     // 0..7 : 512-column slab
  const int tid   = threadIdx.x;

  // stage emb rows for this b: 512 uint4 of data + 32 uint4 of zeros
  const uint4* src = (const uint4*)(emb_bf16 + (size_t)b * T_ * D_);
  lds[tid * 2]     = src[tid * 2];
  lds[tid * 2 + 1] = src[tid * 2 + 1];
  if (tid < 32) lds[512 + tid] = make_uint4(0u, 0u, 0u, 0u);
  __syncthreads();

  const __hip_bfloat16* lds_bf = (const __hip_bfloat16*)lds;

  // Wave-uniform quantities -> SGPRs (readfirstlane): scalar loop control.
  const int wave = __builtin_amdgcn_readfirstlane(tid >> 5);
  const int lane = tid & 31;
  const int n0   = xslab * 512 + wave * 64;         // SGPR: first of 64 columns
  const int l    = n0 >> 8;                         // SGPR: l-block of this wave

  // A-fragment addressing (16x32 bf16): lane&15 = row t, lane>>4 picks K half
  const int tl   = lane & 15;
  const int koff = (lane >> 4) * 16;
  // B-fragment addressing (32x16 bf16): lane&15 = column, 16 contiguous K per lane
  const int kb   = (lane >> 4) * 16;

  // Flattened K step s = m*8 + (d0/32): B element offset = 32*s + kb, so the
  // per-column uint4 pointers below advance by exactly 4 uint4 per step.
  const uint4* pb0 = (const uint4*)(Bt + (size_t)(n0 +  0 + (lane & 15)) * (L_ * D_) + kb);
  const uint4* pb1 = (const uint4*)(Bt + (size_t)(n0 + 16 + (lane & 15)) * (L_ * D_) + kb);
  const uint4* pb2 = (const uint4*)(Bt + (size_t)(n0 + 32 + (lane & 15)) * (L_ * D_) + kb);
  const uint4* pb3 = (const uint4*)(Bt + (size_t)(n0 + 48 + (lane & 15)) * (L_ * D_) + kb);

  v8f acc0 = {}, acc1 = {}, acc2 = {}, acc3 = {};

  auto load_step = [&](int s, Frag32& a, Frag32& f0, Frag32& f1,
                       Frag32& f2, Frag32& f3) {
    const int m    = s >> 3;                        // scalar (s is scalar)
    const int d0   = (s & 7) << 5;
    const int rowA = (tl >= m) ? (tl - m) : 16;     // causal zero-pad -> zero row
    const uint4* pa = (const uint4*)(lds_bf + rowA * D_ + d0 + koff);
    a.u[0]  = pa[0];
    a.u[1]  = pa[1];                                // +8 bf16 elements
    const int ko = s * 4;
    f0.u[0] = pb0[ko]; f0.u[1] = pb0[ko + 1];
    f1.u[0] = pb1[ko]; f1.u[1] = pb1[ko + 1];
    f2.u[0] = pb2[ko]; f2.u[1] = pb2[ko + 1];
    f3.u[0] = pb3[ko]; f3.u[1] = pb3[ko + 1];
  };

  auto do_wmma = [&](const Frag32& a, const Frag32& f0, const Frag32& f1,
                     const Frag32& f2, const Frag32& f3) {
    acc0 = __builtin_amdgcn_wmma_f32_16x16x32_bf16(false, a.v, false, f0.v,
                                                   (short)0, acc0, false, false);
    acc1 = __builtin_amdgcn_wmma_f32_16x16x32_bf16(false, a.v, false, f1.v,
                                                   (short)0, acc1, false, false);
    acc2 = __builtin_amdgcn_wmma_f32_16x16x32_bf16(false, a.v, false, f2.v,
                                                   (short)0, acc2, false, false);
    acc3 = __builtin_amdgcn_wmma_f32_16x16x32_bf16(false, a.v, false, f3.v,
                                                   (short)0, acc3, false, false);
  };

  const int nsteps = (l + 1) * 8;                   // SGPR; triangular skip (m<=l)
  Frag32 aP, p0, p1, p2, p3;                        // ping
  Frag32 aQ, q0, q1, q2, q3;                        // pong
  load_step(0, aP, p0, p1, p2, p3);
  for (int s = 0; s < nsteps; s += 2) {             // scalar branch, nsteps even
    load_step(s + 1, aQ, q0, q1, q2, q3);
    do_wmma(aP, p0, p1, p2, p3);
    if (s + 2 < nsteps)                             // scalar guard
      load_step(s + 2, aP, p0, p1, p2, p3);
    do_wmma(aQ, q0, q1, q2, q3);
  }

  // Epilogue: + conv_b, relu, sigmoid, store bf16 gate f[l][b][t][c]
  const int trow = (lane >> 4) * 8;                 // C layout: VGPR i -> M = i (+8)
#pragma unroll
  for (int j = 0; j < 4; ++j) {
    const v8f acc = (j == 0) ? acc0 : (j == 1) ? acc1 : (j == 2) ? acc2 : acc3;
    const int cin = ((n0 + 16 * j + (lane & 15)) & (D_ - 1));
    const float bias = conv_b[l * D_ + cin];
    __hip_bfloat16* fp = fbuf + (((size_t)l * B_ + b) * T_) * D_ + cin;
#pragma unroll
    for (int i = 0; i < 8; ++i) {
      int t = trow + i;
      float v = acc[i] + bias;
      v = v > 0.f ? v : 0.f;                        // relu
      float g = 1.0f / (1.0f + __expf(-v));         // sigmoid
      fp[(size_t)t * D_] = __float2bfloat16(g);
    }
  }
}

// ---------------------------------------------------------------------------
// K3: fused triple QRNN fo-pool + sum over (l, t) -> o[B, D]
// grid = B, block = D.  h1,h2,h3 carried in registers across t (chain fusion).
// ---------------------------------------------------------------------------
__global__ void __launch_bounds__(256)
scan_kernel(const __hip_bfloat16* __restrict__ fbuf,
            const float* __restrict__ emb_f32,
            float* __restrict__ o) {
  int b = blockIdx.x, c = threadIdx.x;
  float x[T_];
#pragma unroll
  for (int t = 0; t < T_; ++t) x[t] = emb_f32[((size_t)b * T_ + t) * D_ + c];
  float oacc = 0.f;
  for (int l = 0; l < L_; ++l) {
    float h1 = 0.f, h2 = 0.f, h3 = 0.f;
    const __hip_bfloat16* fp = fbuf + (((size_t)l * B_ + b) * T_) * D_ + c;
#pragma unroll
    for (int t = 0; t < T_; ++t) {
      float f = __bfloat162float(fp[(size_t)t * D_]);
      float g = 1.f - f;
      h1 = f * x[t] + g * h1;
      h2 = f * h1   + g * h2;
      h3 = f * h2   + g * h3;
      oacc += h3;
    }
  }
  o[(size_t)b * D_ + c] = oacc;
}

// ---------------------------------------------------------------------------
// K4: head.  z = [o, user_emb[user]] @ fc1_w.T + fc1_b ;
//     res[b,n] = dot(W2[item_var[b,n]], z[b]) + b2[item_var[b,n]]
// grid = B, block = 256.  Wave-level __shfl_xor reduction for the 32 targets.
// ---------------------------------------------------------------------------
__global__ void __launch_bounds__(256)
head_kernel(const float* __restrict__ o,
            const int* __restrict__ user_var,
            const float* __restrict__ user_emb,
            const float* __restrict__ fc1_w,
            const float* __restrict__ fc1_b,
            const int* __restrict__ item_var,
            const float* __restrict__ W2,
            const float* __restrict__ b2,
            float* __restrict__ out) {
  __shared__ float cat[2 * D_];
  __shared__ float z[D_];
  int b = blockIdx.x, tid = threadIdx.x;
  cat[tid]      = o[(size_t)b * D_ + tid];
  cat[D_ + tid] = user_emb[(size_t)user_var[b] * D_ + tid];
  __syncthreads();

  const float* wrow = fc1_w + (size_t)tid * (2 * D_);
  float s = fc1_b[tid];
#pragma unroll 8
  for (int j = 0; j < 2 * D_; ++j) s += cat[j] * wrow[j];
  z[tid] = s;
  __syncthreads();

  int n = tid >> 3, k8 = tid & 7;
  int idx = item_var[b * NTGT_ + n];
  const float* w2r = W2 + (size_t)idx * D_;
  float p = 0.f;
#pragma unroll
  for (int k = k8 * 32; k < k8 * 32 + 32; ++k) p += w2r[k] * z[k];
  p += __shfl_xor(p, 1);
  p += __shfl_xor(p, 2);
  p += __shfl_xor(p, 4);
  if (k8 == 0) out[b * NTGT_ + n] = p + b2[idx];
}

// ---------------------------------------------------------------------------
// Launcher.  Workspace layout (256B-aligned offsets), total ~108.5 MB:
//   [0)            emb_f32   B*T*D f32      8,388,608 B
//   [8388608)      emb_bf16  B*T*D bf16     4,194,304 B
//   [12582912)     Bt        4096x4096 bf16 33,554,432 B
//   [46137344)     fbuf      L*B*T*D bf16   67,108,864 B
//   [113246208)    o         B*D f32           524,288 B
// ---------------------------------------------------------------------------
extern "C" void kernel_launch(void* const* d_in, const int* in_sizes, int n_in,
                              void* d_out, int out_size, void* d_ws, size_t ws_size,
                              hipStream_t stream) {
  const int*   seq_var  = (const int*)d_in[0];
  const int*   user_var = (const int*)d_in[1];
  const int*   item_var = (const int*)d_in[2];
  const float* item_emb = (const float*)d_in[3];
  const float* user_emb = (const float*)d_in[4];
  const float* conv_w   = (const float*)d_in[5];
  const float* conv_b   = (const float*)d_in[6];
  const float* fc1_w    = (const float*)d_in[7];
  const float* fc1_b    = (const float*)d_in[8];
  const float* W2       = (const float*)d_in[9];
  const float* b2       = (const float*)d_in[10];
  float* out = (float*)d_out;

  unsigned char* ws = (unsigned char*)d_ws;
  float*          emb_f32  = (float*)(ws);
  __hip_bfloat16* emb_bf16 = (__hip_bfloat16*)(ws + 8388608);
  __hip_bfloat16* Bt       = (__hip_bfloat16*)(ws + 12582912);
  __hip_bfloat16* fbuf     = (__hip_bfloat16*)(ws + 46137344);
  float*          o        = (float*)(ws + 113246208);

  gather_emb_kernel<<<dim3(B_ * T_), dim3(D_), 0, stream>>>(seq_var, item_emb,
                                                            emb_f32, emb_bf16);
  pack_w_kernel<<<dim3(L_ * L_, D_), dim3(D_), 0, stream>>>(conv_w, Bt);
  conv_gemm_kernel<<<dim3(8, B_), dim3(256), 0, stream>>>(emb_bf16, Bt, conv_b, fbuf);
  scan_kernel<<<dim3(B_), dim3(D_), 0, stream>>>(fbuf, emb_f32, o);
  head_kernel<<<dim3(B_), dim3(D_), 0, stream>>>(o, user_var, user_emb, fc1_w, fc1_b,
                                                 item_var, W2, b2, out);
}
